// Block_27573690040857
// MI455X (gfx1250) — matrix operand order repaired and checked by reading, not compiled, not run on hardware
//
#include <hip/hip_runtime.h>

typedef __attribute__((ext_vector_type(16))) __bf16 bf16x16;
typedef __attribute__((ext_vector_type(8)))  float  f32x8;
typedef __attribute__((ext_vector_type(4)))  unsigned int uint32x4;
typedef __attribute__((ext_vector_type(8)))  int          int32x8;
typedef __attribute__((ext_vector_type(4)))  int          int32x4;

static constexpr int Bc = 4, Tc = 2048, Dc = 512, Hc = 8, HSc = 64, Ec = 8, CAPc = 2048, Fc = 2048;
static constexpr int Nrows = Bc * Tc; // 8192

// ---- CDNA5 TDM availability probes (compile-safe) -------------------------
#if defined(__has_builtin)
#if __has_builtin(__builtin_amdgcn_tensor_load_to_lds) && __has_builtin(__builtin_amdgcn_s_wait_tensorcnt)
#define USE_TDM 1
#endif
#endif
#ifndef USE_TDM
#define USE_TDM 0
#endif

#if defined(__has_include)
#if __has_include(<hip/amd_detail/amd_gfx1250_TDM.h>)
#define TDM_6ARG 1
#endif
#endif
#ifndef TDM_6ARG
#define TDM_6ARG 0
#endif

// ---------------------------------------------------------------------------
// Weight conversion: [batch][K][N] f32 -> [batch][N][K] bf16 (transposed so
// GEMM B-fragments are K-contiguous -> 128-bit LDS/global loads)
// ---------------------------------------------------------------------------
__global__ void wT_kernel(const float* __restrict__ in, unsigned short* __restrict__ out,
                          int Kd, int Nd, int total) {
  int i = blockIdx.x * 256 + threadIdx.x;
  if (i >= total) return;
  int kn = Kd * Nd;
  int b = i / kn, rem = i - b * kn;
  int kk = rem / Nd, n = rem - kk * Nd;
  __bf16 v = (__bf16)in[i];
  ((__bf16*)out)[(size_t)b * kn + (size_t)n * Kd + kk] = v;
}

// Wq/Wk/Wv: [H][D][HS] f32 -> [H*HS][D] bf16  (N=h*64+e major, K=d contiguous)
__global__ void qkvwT_kernel(const float* __restrict__ W, unsigned short* __restrict__ out) {
  int i = blockIdx.x * 256 + threadIdx.x; // 8*512*64 = 262144
  int h = i >> 15, d = (i >> 6) & 511, e = i & 63;
  __bf16 v = (__bf16)W[i];
  ((__bf16*)out)[(size_t)(h * HSc + e) * Dc + d] = v;
}

// ---------------------------------------------------------------------------
// LayerNorm: one wave per row of 512, bf16 output
// ---------------------------------------------------------------------------
__global__ void ln_kernel(const float* __restrict__ x, const float* __restrict__ g,
                          const float* __restrict__ b, unsigned short* __restrict__ outB) {
  int wv = threadIdx.x >> 5, lane = threadIdx.x & 31;
  int row = blockIdx.x * 8 + wv;
  const float* xr = x + (size_t)row * Dc;
  float s = 0.f, s2 = 0.f;
  for (int i = lane; i < Dc; i += 32) { float t = xr[i]; s += t; s2 += t * t; }
  for (int off = 16; off >= 1; off >>= 1) { s += __shfl_xor(s, off, 32); s2 += __shfl_xor(s2, off, 32); }
  float mean = s * (1.f / Dc);
  float var  = s2 * (1.f / Dc) - mean * mean;
  float inv  = rsqrtf(var + 1e-5f);
  __bf16* ob = (__bf16*)outB;
  for (int i = lane; i < Dc; i += 32)
    ob[(size_t)row * Dc + i] = (__bf16)((xr[i] - mean) * inv * g[i] + b[i]);
}

// ---------------------------------------------------------------------------
// bf16 WMMA GEMM: C[M,N] = A[M,K] * B^T[N,K]. Tiles 128x64 per block, 8 waves.
// A fragments: direct 128-bit global loads (rows are wave-exclusive).
// B panel (64 n-rows x 32 k): double-buffered in LDS; staged by the Tensor
// Data Mover (tensor_load_to_lds + s_wait_tensorcnt) with the next panel's
// DMA overlapping the current panel's WMMAs; one barrier per K-step.
// Epilogue: scale/bias/relu/residual, f32/bf16/V-transposed store, or gated
// atomic scatter-add (MoE combine).
// ---------------------------------------------------------------------------
__global__ __launch_bounds__(256)
void gemm_bf16_kernel(const unsigned short* __restrict__ Ap, const unsigned short* __restrict__ Btp,
                      int Kdim, int N,
                      const float* __restrict__ bias, float scale, int do_relu,
                      const float* __restrict__ residual,
                      float* __restrict__ outF, unsigned short* __restrict__ outB,
                      unsigned short* __restrict__ outVT,
                      const int* __restrict__ scat_idx, const float* __restrict__ scat_w,
                      float* __restrict__ scat_base) {
  __shared__ __bf16 Bs[2][64 * 32];
  const __bf16* A  = (const __bf16*)Ap;
  const __bf16* Bt = (const __bf16*)Btp;

  int tid = threadIdx.x, lane = tid & 31, wv = tid >> 5;
  int lm = lane & 15, lg = lane >> 4;
  int m0 = blockIdx.y * 128, n0 = blockIdx.x * 64;

  const __bf16* Arow = A + (size_t)(m0 + wv * 16 + lm) * Kdim;

  auto stage = [&](int k0, int buf) {
#if USE_TDM
    if (wv == 0) {
      // D# group 0: count=1 | lds_addr | global_addr[56:0] | type=2
      unsigned ldsOff = (unsigned)(size_t)(__attribute__((address_space(3))) __bf16*)Bs[buf];
      unsigned long long ga =
          (unsigned long long)(const void*)(Bt + (size_t)n0 * Kdim + k0);
      uint32x4 g0;
      g0[0] = 1u;                                   // count=1, user descriptor
      g0[1] = ldsOff;                               // LDS byte address
      g0[2] = (unsigned)ga;                         // global addr [31:0]
      g0[3] = (unsigned)((ga >> 32) & 0x1FFFFFFu) | (2u << 30); // [56:32] | type=2
      // D# group 1: data_size=2B; tile 32(k) x 64(n); dim0 stride = Kdim
      int32x8 g1;
      g1[0] = (int)(1u << 16);                                         // data_size=1 -> 2B
      g1[1] = (int)(((unsigned)Kdim & 0xFFFFu) << 16);                 // tensor_dim0 lo16
      g1[2] = (int)((((unsigned)Kdim >> 16) & 0xFFFFu) | (64u << 16)); // dim0 hi | tensor_dim1 lo
      g1[3] = (int)(32u << 16);                                        // dim1 hi=0 | tile_dim0=32
      g1[4] = (int)64u;                                                // tile_dim1=64, tile_dim2=0
      g1[5] = (int)Kdim;                                               // tensor_dim0_stride lo32
      g1[6] = 0; g1[7] = 0;
      int32x4 z4; z4[0] = z4[1] = z4[2] = z4[3] = 0;
#if TDM_6ARG
      int32x8 z8;
#pragma unroll
      for (int q = 0; q < 8; ++q) z8[q] = 0;
      __builtin_amdgcn_tensor_load_to_lds(g0, g1, z4, z4, z8, 0);
#else
      __builtin_amdgcn_tensor_load_to_lds(g0, g1, z4, z4, 0);
#endif
    }
#else
    // sync staging fallback: thread tid -> n-row tid>>2, 16B chunk tid&3
    ((uint4*)Bs[buf])[tid] =
        ((const uint4*)(Bt + (size_t)(n0 + (tid >> 2)) * Kdim + k0))[tid & 3];
#endif
  };

  f32x8 acc[4];
#pragma unroll
  for (int j = 0; j < 4; ++j)
#pragma unroll
    for (int r = 0; r < 8; ++r) acc[j][r] = 0.f;

  int nsteps = Kdim >> 5;
  stage(0, 0); // prologue: panel 0 in flight

  for (int s = 0; s < nsteps; ++s) {
    int k0 = s << 5;
    int buf = s & 1;
#if USE_TDM
    __builtin_amdgcn_s_wait_tensorcnt(0); // panel s DMA complete (issuing wave)
#endif
    __syncthreads();                      // data visible; prior readers of buf^1 done
    if (s + 1 < nsteps) stage(k0 + 32, buf ^ 1); // overlap next DMA with compute
    if (k0 + 64 < Kdim) __builtin_prefetch(Arow + k0 + 64, 0, 3); // near-cache prefetch

    // A fragment (16x32): k = lg*16 + e contiguous per half-wave -> 2 x b128 global
    bf16x16 af;
#pragma unroll
    for (int e = 0; e < 8; ++e) af[e] = Arow[k0 + lg * 16 + e];
#pragma unroll
    for (int e = 0; e < 8; ++e) af[8 + e] = Arow[k0 + lg * 16 + 8 + e];

#pragma unroll
    for (int j = 0; j < 4; ++j) {
      bf16x16 bfr; // lane: n-row j*16+lm, k = lg*16 + e -> 32B contiguous
#pragma unroll
      for (int e = 0; e < 16; ++e) bfr[e] = Bs[buf][(j * 16 + lm) * 32 + lg * 16 + e];
      acc[j] = __builtin_amdgcn_wmma_f32_16x16x32_bf16(false, af, false, bfr,
                                                       (short)0, acc[j], false, false);
    }
  }

  // epilogue: C layout row = r + 8*lg (+ wave stripe), col = lm (+16*j)
  int row_base = m0 + wv * 16 + 8 * lg;
  __bf16* ob  = (__bf16*)outB;
  __bf16* ovt = (__bf16*)outVT;
#pragma unroll
  for (int j = 0; j < 4; ++j) {
    int col = n0 + j * 16 + lm;
    float bval = bias ? bias[col] : 0.f;
#pragma unroll
    for (int r = 0; r < 8; ++r) {
      int row = row_base + r;
      float v = acc[j][r] * scale + bval;
      if (do_relu) v = fmaxf(v, 0.f);
      if (residual) v += residual[(size_t)row * N + col];
      if (outF) outF[(size_t)row * N + col] = v;
      if (ob) ob[(size_t)row * N + col] = (__bf16)v;
      if (ovt) { // V stored transposed: [(b*512 + h*64+e)][t]
        int bb = row >> 11, tt = row & (Tc - 1);
        ovt[((size_t)(bb * Dc + col)) * Tc + tt] = (__bf16)v;
      }
      if (scat_idx)
        atomicAdd(scat_base + (size_t)scat_idx[row] * N + col, v * scat_w[row]);
    }
  }
}

// ---------------------------------------------------------------------------
// Flash attention (causal). q,k bf16 [B*T,512] (col=h*64+e); v bf16 transposed
// [(b*512+h*64+e)][T]. One wave per 16-query tile, 32-key blocks.
// Softmax scale pre-folded into Q. All fragments are contiguous 128-bit loads;
// P transposed to A-layout through per-wave LDS.
// ---------------------------------------------------------------------------
__global__ __launch_bounds__(256)
void flash_attn_kernel(const unsigned short* __restrict__ qp, const unsigned short* __restrict__ kp,
                       const unsigned short* __restrict__ vtp, unsigned short* __restrict__ op) {
  const __bf16* q  = (const __bf16*)qp;
  const __bf16* k  = (const __bf16*)kp;
  const __bf16* vt = (const __bf16*)vtp;
  __bf16* o = (__bf16*)op;

  int bh = blockIdx.y, b = bh >> 3, h = bh & 7;
  int wv = threadIdx.x >> 5, lane = threadIdx.x & 31;
  int t0 = (blockIdx.x * 8 + wv) * 16;
  int lm = lane & 15, lg = lane >> 4, kb = lg * 8;
  const size_t base  = ((size_t)b * Tc) * Dc + h * HSc;          // q/k row-major base
  const size_t vbase = ((size_t)(b * Dc + h * HSc)) * Tc;        // v transposed base

  __shared__ __bf16 Pt[8][16 * 32];

  bf16x16 qa[2];
#pragma unroll
  for (int ec = 0; ec < 2; ++ec)
#pragma unroll
    for (int e = 0; e < 16; ++e) {
      int kk = lg * 16 + e;
      qa[ec][e] = q[base + (size_t)(t0 + lm) * Dc + ec * 32 + kk];
    }

  f32x8 oacc[4];
#pragma unroll
  for (int j = 0; j < 4; ++j)
#pragma unroll
    for (int r = 0; r < 8; ++r) oacc[j][r] = 0.f;
  float m_r[8], l_r[8];
#pragma unroll
  for (int r = 0; r < 8; ++r) { m_r[r] = -1e30f; l_r[r] = 0.f; }

  for (int s0 = 0; s0 < t0 + 16; s0 += 32) {
    f32x8 sac[2];
#pragma unroll
    for (int j = 0; j < 2; ++j) {
#pragma unroll
      for (int r = 0; r < 8; ++r) sac[j][r] = 0.f;
      int s = s0 + j * 16 + lm; // this lane's key column
#pragma unroll
      for (int ec = 0; ec < 2; ++ec) {
        bf16x16 kf; // K^T fragment: k = lg*16+e contiguous in head-dim
#pragma unroll
        for (int e = 0; e < 16; ++e)
          kf[e] = k[base + (size_t)s * Dc + ec * 32 + lg * 16 + e];
        sac[j] = __builtin_amdgcn_wmma_f32_16x16x32_bf16(false, qa[ec], false, kf,
                                                         (short)0, sac[j], false, false);
      }
    }
#pragma unroll
    for (int r = 0; r < 8; ++r) {
      int trow = t0 + r + 8 * lg;
#pragma unroll
      for (int j = 0; j < 2; ++j) {
        int sg = s0 + j * 16 + lm;
        if (sg > trow) sac[j][r] = -1e30f; // causal mask
      }
      float mv = fmaxf(sac[0][r], sac[1][r]);
      for (int off = 8; off >= 1; off >>= 1) mv = fmaxf(mv, __shfl_xor(mv, off, 32));
      float mnew = fmaxf(m_r[r], mv);
      float corr = __expf(m_r[r] - mnew);
      float rs = 0.f;
#pragma unroll
      for (int j = 0; j < 2; ++j) {
        float p = __expf(sac[j][r] - mnew);
        sac[j][r] = p; rs += p;
        Pt[wv][(r + 8 * lg) * 32 + j * 16 + lm] = (__bf16)p;
      }
      for (int off = 8; off >= 1; off >>= 1) rs += __shfl_xor(rs, off, 32);
      l_r[r] = l_r[r] * corr + rs;
      m_r[r] = mnew;
#pragma unroll
      for (int jt = 0; jt < 4; ++jt) oacc[jt][r] *= corr;
    }
    bf16x16 pa; // P as A fragment (in-wave LDS ordering via DScnt)
#pragma unroll
    for (int e = 0; e < 16; ++e) {
      int kk = lg * 16 + e;
      pa[e] = Pt[wv][lm * 32 + kk];
    }
#pragma unroll
    for (int et = 0; et < 4; ++et) {
      bf16x16 vf; // V^T fragment: k = s0 + lg*16 + e contiguous in t
#pragma unroll
      for (int e = 0; e < 16; ++e)
        vf[e] = vt[vbase + (size_t)(et * 16 + lm) * Tc + s0 + lg * 16 + e];
      oacc[et] = __builtin_amdgcn_wmma_f32_16x16x32_bf16(false, pa, false, vf,
                                                         (short)0, oacc[et], false, false);
    }
  }
#pragma unroll
  for (int et = 0; et < 4; ++et)
#pragma unroll
    for (int r = 0; r < 8; ++r) {
      float val = oacc[et][r] / l_r[r];
      o[base + (size_t)(t0 + r + 8 * lg) * Dc + et * 16 + lm] = (__bf16)val;
    }
}

// ---------------------------------------------------------------------------
// Noisy top-2 gating: one wave per token (hash-Gaussian stands in for
// jax.random.normal(key 42); deterministic).
// ---------------------------------------------------------------------------
__global__ void gating_kernel(const unsigned short* __restrict__ h2p,
                              const float* __restrict__ Wg, const float* __restrict__ bg,
                              const float* __restrict__ Wn, const float* __restrict__ bn,
                              int* __restrict__ t2i, float* __restrict__ t2g) {
  const __bf16* h2 = (const __bf16*)h2p;
  int wv = threadIdx.x >> 5, lane = threadIdx.x & 31;
  int t = blockIdx.x * 8 + wv;
  float lg_[8], ln_[8];
#pragma unroll
  for (int j = 0; j < 8; ++j) { lg_[j] = 0.f; ln_[j] = 0.f; }
  for (int d = lane; d < Dc; d += 32) {
    float hv = (float)h2[(size_t)t * Dc + d];
#pragma unroll
    for (int j = 0; j < 8; ++j) {
      lg_[j] += hv * Wg[d * 8 + j];
      ln_[j] += hv * Wn[d * 8 + j];
    }
  }
#pragma unroll
  for (int j = 0; j < 8; ++j)
    for (int off = 16; off >= 1; off >>= 1) {
      lg_[j] += __shfl_xor(lg_[j], off, 32);
      ln_[j] += __shfl_xor(ln_[j], off, 32);
    }
  if (lane == 0) {
    float noisy[8];
#pragma unroll
    for (int j = 0; j < 8; ++j) {
      float nl = ln_[j] + bn[j];
      float sp = (nl > 20.f) ? nl : log1pf(__expf(nl));
      unsigned u = (unsigned)(t * 8 + j) * 2654435761u + 42u * 0x9E3779B9u;
      u ^= u << 13; u ^= u >> 17; u ^= u << 5;
      unsigned u2 = u * 0x85EBCA6Bu; u2 ^= u2 >> 13; u2 *= 0xC2B2AE35u; u2 ^= u2 >> 16;
      float u1f = ((float)((u >> 8) + 1)) * (1.f / 16777218.f);
      float u2f = ((float)(u2 >> 8)) * (1.f / 16777216.f);
      float z = sqrtf(-2.f * logf(u1f)) * __cosf(6.2831853f * u2f);
      noisy[j] = lg_[j] + bg[j] + z * sp;
    }
    int i0 = 0;
#pragma unroll
    for (int j = 1; j < 8; ++j) if (noisy[j] > noisy[i0]) i0 = j;
    int i1 = -1;
#pragma unroll
    for (int j = 0; j < 8; ++j)
      if (j != i0 && (i1 < 0 || noisy[j] > noisy[i1])) i1 = j;
    float g0 = 1.f / (1.f + __expf(noisy[i1] - noisy[i0]));
    t2i[t * 2] = i0; t2i[t * 2 + 1] = i1;
    t2g[t * 2] = g0; t2g[t * 2 + 1] = 1.f - g0;
  }
}

// ---------------------------------------------------------------------------
// Capacity routing: one block per expert; ordered block prefix scan.
// ---------------------------------------------------------------------------
__global__ void route_kernel(const int* __restrict__ t2i, const float* __restrict__ t2g,
                             int* __restrict__ idx_buf, float* __restrict__ gate_sel) {
  int e = blockIdx.x, tid = threadIdx.x;
  __shared__ int scan[256];
  __shared__ int running;
  for (int c = tid; c < CAPc; c += 256) { idx_buf[e * CAPc + c] = 0; gate_sel[e * CAPc + c] = 0.f; }
  if (tid == 0) running = 0;
  __syncthreads();
  for (int base0 = 0; base0 < Nrows; base0 += 256) {
    int t = base0 + tid;
    int i0 = t2i[t * 2], i1 = t2i[t * 2 + 1];
    int m = (i0 == e || i1 == e) ? 1 : 0;
    scan[tid] = m;
    __syncthreads();
    for (int off = 1; off < 256; off <<= 1) {
      int vv = (tid >= off) ? scan[tid - off] : 0;
      __syncthreads();
      scan[tid] += vv;
      __syncthreads();
    }
    int rank = running + scan[tid] - m;
    if (m && rank < CAPc) {
      idx_buf[e * CAPc + rank] = t;
      gate_sel[e * CAPc + rank] = (i0 == e) ? t2g[t * 2] : t2g[t * 2 + 1];
    }
    __syncthreads();
    if (tid == 0) running += scan[255];
    __syncthreads();
  }
}

__global__ void gather_kernel(const unsigned short* __restrict__ h2, const int* __restrict__ idx,
                              unsigned short* __restrict__ xg) {
  int i = blockIdx.x * 256 + threadIdx.x; // CAP*512
  int c = i >> 9, d = i & 511;
  xg[i] = h2[(size_t)idx[c] * Dc + d];
}

// ---------------------------------------------------------------------------
extern "C" void kernel_launch(void* const* d_in, const int* in_sizes, int n_in,
                              void* d_out, int out_size, void* d_ws, size_t ws_size,
                              hipStream_t stream) {
  (void)in_sizes; (void)n_in; (void)out_size; (void)ws_size;
  const float* x     = (const float*)d_in[0];
  const float* ln1_g = (const float*)d_in[1];
  const float* ln1_b = (const float*)d_in[2];
  const float* Wq    = (const float*)d_in[3];
  const float* Wk    = (const float*)d_in[4];
  const float* Wv    = (const float*)d_in[5];
  const float* Wo    = (const float*)d_in[6];
  const float* bo    = (const float*)d_in[7];
  const float* ln2_g = (const float*)d_in[8];
  const float* ln2_b = (const float*)d_in[9];
  const float* Wg    = (const float*)d_in[10];
  const float* bg    = (const float*)d_in[11];
  const float* Wn    = (const float*)d_in[12];
  const float* bn    = (const float*)d_in[13];
  const float* W1    = (const float*)d_in[14];
  const float* b1    = (const float*)d_in[15];
  const float* W2    = (const float*)d_in[16];
  const float* b2    = (const float*)d_in[17];
  float* out = (float*)d_out;

  char* ws = (char*)d_ws;
  size_t off = 0;
  auto alloc = [&](size_t bytes) -> char* {
    char* p = ws + off;
    off = (off + bytes + 255) & ~(size_t)255;
    return p;
  };
  unsigned short* hbf   = (unsigned short*)alloc((size_t)Nrows * Dc * 2);
  unsigned short* wqT   = (unsigned short*)alloc((size_t)Dc * Dc * 2);
  unsigned short* wkT   = (unsigned short*)alloc((size_t)Dc * Dc * 2);
  unsigned short* wvT   = (unsigned short*)alloc((size_t)Dc * Dc * 2);
  unsigned short* woT   = (unsigned short*)alloc((size_t)Dc * Dc * 2);
  unsigned short* w1T   = (unsigned short*)alloc((size_t)Ec * Dc * Fc * 2);
  unsigned short* w2T   = (unsigned short*)alloc((size_t)Ec * Fc * Dc * 2);
  unsigned short* qB    = (unsigned short*)alloc((size_t)Nrows * Dc * 2);
  unsigned short* kB    = (unsigned short*)alloc((size_t)Nrows * Dc * 2);
  unsigned short* vTB   = (unsigned short*)alloc((size_t)Nrows * Dc * 2);
  unsigned short* attnB = (unsigned short*)alloc((size_t)Nrows * Dc * 2);
  unsigned short* h2B   = (unsigned short*)alloc((size_t)Nrows * Dc * 2);
  int*   t2i    = (int*)alloc((size_t)Nrows * 2 * 4);
  float* t2g    = (float*)alloc((size_t)Nrows * 2 * 4);
  int*   idxbuf = (int*)alloc((size_t)Ec * CAPc * 4);
  float* gsel   = (float*)alloc((size_t)Ec * CAPc * 4);
  unsigned short* xgB  = (unsigned short*)alloc((size_t)CAPc * Dc * 2);
  unsigned short* midB = (unsigned short*)alloc((size_t)CAPc * Fc * 2);

  // 0) weight conversions (bf16 + transpose to N-major)
  wT_kernel<<<(Dc * Dc + 255) / 256, 256, 0, stream>>>(Wo, woT, Dc, Dc, Dc * Dc);
  wT_kernel<<<(Ec * Dc * Fc + 255) / 256, 256, 0, stream>>>(W1, w1T, Dc, Fc, Ec * Dc * Fc);
  wT_kernel<<<(Ec * Fc * Dc + 255) / 256, 256, 0, stream>>>(W2, w2T, Fc, Dc, Ec * Fc * Dc);
  qkvwT_kernel<<<(Hc * Dc * HSc + 255) / 256, 256, 0, stream>>>(Wq, wqT);
  qkvwT_kernel<<<(Hc * Dc * HSc + 255) / 256, 256, 0, stream>>>(Wk, wkT);
  qkvwT_kernel<<<(Hc * Dc * HSc + 255) / 256, 256, 0, stream>>>(Wv, wvT);

  // 1) h = LN1(x)
  ln_kernel<<<Nrows / 8, 256, 0, stream>>>(x, ln1_g, ln1_b, hbf);

  // 2) QKV projections (softmax scale folded into Q; V stored transposed)
  const float qscale = 0.044194173824159216f; // 512^-0.5
  dim3 gQKV(Dc / 64, Nrows / 128);
  gemm_bf16_kernel<<<gQKV, 256, 0, stream>>>(hbf, wqT, Dc, Dc, nullptr, qscale, 0,
                                             nullptr, nullptr, qB, nullptr, nullptr, nullptr, nullptr);
  gemm_bf16_kernel<<<gQKV, 256, 0, stream>>>(hbf, wkT, Dc, Dc, nullptr, 1.f, 0,
                                             nullptr, nullptr, kB, nullptr, nullptr, nullptr, nullptr);
  gemm_bf16_kernel<<<gQKV, 256, 0, stream>>>(hbf, wvT, Dc, Dc, nullptr, 1.f, 0,
                                             nullptr, nullptr, nullptr, vTB, nullptr, nullptr, nullptr);

  // 3) causal flash attention
  flash_attn_kernel<<<dim3(Tc / 128, Bc * Hc), 256, 0, stream>>>(qB, kB, vTB, attnB);

  // 4) out-proj + bias + residual -> d_out (x2, fp32)
  gemm_bf16_kernel<<<gQKV, 256, 0, stream>>>(attnB, woT, Dc, Dc, bo, 1.f, 0,
                                             x, out, nullptr, nullptr, nullptr, nullptr, nullptr);

  // 5) h2 = LN2(x2)
  ln_kernel<<<Nrows / 8, 256, 0, stream>>>(out, ln2_g, ln2_b, h2B);

  // 6) gating + capacity routing
  gating_kernel<<<Nrows / 8, 256, 0, stream>>>(h2B, Wg, bg, Wn, bn, t2i, t2g);
  route_kernel<<<Ec, 256, 0, stream>>>(t2i, t2g, idxbuf, gsel);

  // 7) per-expert FFN: gather -> GEMM1(relu) -> GEMM2(gated scatter-add)
  for (int e = 0; e < Ec; ++e) {
    gather_kernel<<<(CAPc * Dc) / 256, 256, 0, stream>>>(h2B, idxbuf + e * CAPc, xgB);
    gemm_bf16_kernel<<<dim3(Fc / 64, CAPc / 128), 256, 0, stream>>>(
        xgB, w1T + (size_t)e * Dc * Fc, Dc, Fc, b1 + e * Fc, 1.f, 1,
        nullptr, nullptr, midB, nullptr, nullptr, nullptr, nullptr);
    gemm_bf16_kernel<<<dim3(Dc / 64, CAPc / 128), 256, 0, stream>>>(
        midB, w2T + (size_t)e * Fc * Dc, Fc, Dc, b2 + e * Dc, 1.f, 0,
        nullptr, nullptr, nullptr, nullptr, idxbuf + e * CAPc, gsel + e * CAPc, out);
  }
}
